// S4Block_16501264351270
// MI455X (gfx1250) — matrix operand order repaired and checked by reading, not compiled, not run on hardware
//
#include <hip/hip_runtime.h>
#include <hip/hip_bf16.h>

// ---------------------------------------------------------------------------
// S4 block on gfx1250: async-to-LDS double-buffered bf16 WMMA gate GEMMs
// (M64xN64 block tile, 2 M-subtiles per wave), chunked parallel linear
// recurrence, LayerNorm.  Shapes: B=8, S=2048, D=1024, M = B*S = 16384.
// ---------------------------------------------------------------------------

typedef __attribute__((ext_vector_type(16))) __bf16 v16bf;
typedef __attribute__((ext_vector_type(8)))  float  v8f;

#define DIM    1024
#define MROWS  16384   // B*S
#define SEQ    2048
#define BATCH  8
#define CHUNK  32      // chunks per sequence
#define CLEN   64      // SEQ / CHUNK

// ---- fp32 -> bf16 round-to-nearest-even ------------------------------------
__device__ __forceinline__ unsigned short f2bf(float f) {
    unsigned int u = __float_as_uint(f);
    u += 0x7FFFu + ((u >> 16) & 1u);
    return (unsigned short)(u >> 16);
}
__device__ __forceinline__ float sigm(float x) {
    return 1.0f / (1.0f + __expf(-x));
}
__device__ __forceinline__ float tanh_fast(float x) {
    float e = __expf(-2.0f * x);
    return (1.0f - e) / (1.0f + e);
}

// ---- LDS helpers: 32-bit LDS offset + async global->LDS copies -------------
__device__ __forceinline__ unsigned to_lds(const void* p) {
    typedef __attribute__((address_space(3))) const char* lds_ptr_t;
    return (unsigned)(unsigned long long)(lds_ptr_t)p;
}
__device__ __forceinline__ void async_ld_b64(unsigned lds, const void* g) {
    asm volatile("global_load_async_to_lds_b64 %0, %1, off"
                 :: "v"(lds), "v"((unsigned long long)g) : "memory");
}
__device__ __forceinline__ void async_ld_b128(unsigned lds, const void* g) {
    asm volatile("global_load_async_to_lds_b128 %0, %1, off"
                 :: "v"(lds), "v"((unsigned long long)g) : "memory");
}

// ---- 1) x: fp32 -> bf16 (vector x4) ----------------------------------------
__global__ void conv_x_kernel(const float* __restrict__ x,
                              unsigned short* __restrict__ xb) {
    int i = blockIdx.x * blockDim.x + threadIdx.x;       // 0 .. M*D/4-1
    float4 v = ((const float4*)x)[i];
    ushort4 o;
    o.x = f2bf(v.x); o.y = f2bf(v.y); o.z = f2bf(v.z); o.w = f2bf(v.w);
    ((ushort4*)xb)[i] = o;
}

// ---- 2) W (K x N, row major fp32) -> W^T bf16 (N-major) via LDS tile -------
__global__ void transpose_conv_w_kernel(const float* __restrict__ W0,
                                        const float* __restrict__ W1,
                                        const float* __restrict__ W2,
                                        const float* __restrict__ W3,
                                        unsigned short* __restrict__ T0,
                                        unsigned short* __restrict__ T1,
                                        unsigned short* __restrict__ T2,
                                        unsigned short* __restrict__ T3) {
    const float* W;
    unsigned short* T;
    switch (blockIdx.z) {
        case 0:  W = W0; T = T0; break;
        case 1:  W = W1; T = T1; break;
        case 2:  W = W2; T = T2; break;
        default: W = W3; T = T3; break;
    }
    __shared__ unsigned short tile[32][33];
    const int tx = threadIdx.x;          // 0..31
    const int ty = threadIdx.y;          // 0..7
    const int n0 = blockIdx.x * 32;
    const int k0 = blockIdx.y * 32;
    #pragma unroll
    for (int r = 0; r < 4; ++r) {
        int k = ty + 8 * r;
        tile[k][tx] = f2bf(W[(size_t)(k0 + k) * DIM + (n0 + tx)]); // coalesced read
    }
    __syncthreads();
    #pragma unroll
    for (int r = 0; r < 4; ++r) {
        int n = ty + 8 * r;
        T[(size_t)(n0 + n) * DIM + (k0 + tx)] = tile[tx][n];       // coalesced write
    }
}

// ---- 3) fused 4-gate GEMM, async-LDS double buffered -----------------------
// grid = (DIM/64, MROWS/64), block = 256 (8 waves: 2 in M x 4 in N).
// Block tile: M=64, N=64, K-step 32. Each wave owns two 16-row M-subtiles
// sharing the same B fragments -> 8 WMMAs per 12 ds_load_b128.
// Per k-stage each thread issues exactly 6 async ops (2xb64 A + 4xb128 B),
// so steady-state wait is s_wait_asynccnt 0x6.
union FragAB { uint4 u[2]; v16bf v; };

#define APAD 40   // padded row length (bf16 elems), 16B-aligned, bank spread

__global__ void __launch_bounds__(256)
gates_gemm_kernel(const unsigned short* __restrict__ xb,
                  const unsigned short* __restrict__ Wti,
                  const unsigned short* __restrict__ Wtf,
                  const unsigned short* __restrict__ Wto,
                  const unsigned short* __restrict__ Wts,
                  const float* __restrict__ bi, const float* __restrict__ bf_,
                  const float* __restrict__ bo, const float* __restrict__ bs,
                  float* __restrict__ Fg,  // sigmoid(f)
                  float* __restrict__ Ag,  // sigmoid(i)*tanh(s)
                  float* __restrict__ Og)  // sigmoid(o)
{
    __shared__ unsigned short As[2][64][APAD];      // 64 x 32 A tile (x2 buf)
    __shared__ unsigned short Bs[2][4][64][APAD];   // 4 gates x 64 x 32 B^T tile

    const int t    = threadIdx.x;
    const int lane = t & 31;
    const int wid  = t >> 5;
    const int wm   = wid & 1;            // 0..1   (M half of block)
    const int wn   = wid >> 1;           // 0..3   (N within block)
    const int half = lane >> 4;          // lane group
    const int l15  = lane & 15;

    const int m0 = blockIdx.y * 64;
    const int n0 = blockIdx.x * 64;

    const unsigned short* xbB = xb  + (size_t)m0 * DIM;
    const unsigned short* wB0 = Wti + (size_t)n0 * DIM;
    const unsigned short* wB1 = Wtf + (size_t)n0 * DIM;
    const unsigned short* wB2 = Wto + (size_t)n0 * DIM;
    const unsigned short* wB3 = Wts + (size_t)n0 * DIM;

    // staging maps:
    //   A: 64x32 = 4KB; thread t copies 2 x 8B: q = t + 256*j -> row q>>3, col (q&7)*4
    //   B: per gate 64x32 = 4KB; thread t copies 16B at row t>>2, seg (t&3)*8
    const int ar0 = t >> 3,          ac0 = (t & 7) * 4;
    const int ar1 = (t + 256) >> 3,  ac1 = ac0;          // rows 32..63
    const int bn  = t >> 2,          bsg = (t & 3) * 8;

    auto issue_stage = [&](int buf, int kk) {
        async_ld_b64 (to_lds(&As[buf][ar0][ac0]),
                      xbB + (size_t)ar0 * DIM + kk + ac0);
        async_ld_b64 (to_lds(&As[buf][ar1][ac1]),
                      xbB + (size_t)ar1 * DIM + kk + ac1);
        async_ld_b128(to_lds(&Bs[buf][0][bn][bsg]),
                      wB0 + (size_t)bn * DIM + kk + bsg);
        async_ld_b128(to_lds(&Bs[buf][1][bn][bsg]),
                      wB1 + (size_t)bn * DIM + kk + bsg);
        async_ld_b128(to_lds(&Bs[buf][2][bn][bsg]),
                      wB2 + (size_t)bn * DIM + kk + bsg);
        async_ld_b128(to_lds(&Bs[buf][3][bn][bsg]),
                      wB3 + (size_t)bn * DIM + kk + bsg);
    };

    // accumulators: 4 gates x 2 M-subtiles
    v8f ci0 = {}, cf0 = {}, co0 = {}, cs0 = {};
    v8f ci1 = {}, cf1 = {}, co1 = {}, cs1 = {};

    auto compute_stage = [&](int buf) {
        // A fragments for the wave's two 16-row subtiles
        FragAB a0, a1;
        const unsigned short* arow0 = &As[buf][wm * 32 + l15][0];
        const unsigned short* arow1 = &As[buf][wm * 32 + 16 + l15][0];
        a0.u[0] = *(const uint4*)(arow0 +      8 * half);
        a0.u[1] = *(const uint4*)(arow0 + 16 + 8 * half);
        a1.u[0] = *(const uint4*)(arow1 +      8 * half);
        a1.u[1] = *(const uint4*)(arow1 + 16 + 8 * half);

        FragAB b;
        const unsigned short* brow;
        brow = &Bs[buf][0][wn * 16 + l15][8 * half];
        b.u[0] = *(const uint4*)brow; b.u[1] = *(const uint4*)(brow + 16);
        ci0 = __builtin_amdgcn_wmma_f32_16x16x32_bf16(false, a0.v, false, b.v,
                                                      (short)0, ci0, false, false);
        ci1 = __builtin_amdgcn_wmma_f32_16x16x32_bf16(false, a1.v, false, b.v,
                                                      (short)0, ci1, false, false);
        brow = &Bs[buf][1][wn * 16 + l15][8 * half];
        b.u[0] = *(const uint4*)brow; b.u[1] = *(const uint4*)(brow + 16);
        cf0 = __builtin_amdgcn_wmma_f32_16x16x32_bf16(false, a0.v, false, b.v,
                                                      (short)0, cf0, false, false);
        cf1 = __builtin_amdgcn_wmma_f32_16x16x32_bf16(false, a1.v, false, b.v,
                                                      (short)0, cf1, false, false);
        brow = &Bs[buf][2][wn * 16 + l15][8 * half];
        b.u[0] = *(const uint4*)brow; b.u[1] = *(const uint4*)(brow + 16);
        co0 = __builtin_amdgcn_wmma_f32_16x16x32_bf16(false, a0.v, false, b.v,
                                                      (short)0, co0, false, false);
        co1 = __builtin_amdgcn_wmma_f32_16x16x32_bf16(false, a1.v, false, b.v,
                                                      (short)0, co1, false, false);
        brow = &Bs[buf][3][wn * 16 + l15][8 * half];
        b.u[0] = *(const uint4*)brow; b.u[1] = *(const uint4*)(brow + 16);
        cs0 = __builtin_amdgcn_wmma_f32_16x16x32_bf16(false, a0.v, false, b.v,
                                                      (short)0, cs0, false, false);
        cs1 = __builtin_amdgcn_wmma_f32_16x16x32_bf16(false, a1.v, false, b.v,
                                                      (short)0, cs1, false, false);
    };

    // ---- software pipeline: last iteration peeled (no in-loop branches) ----
    issue_stage(0, 0);
    int ibuf = 0;
    for (int kk = 0; kk < DIM - 32; kk += 32, ibuf ^= 1) {
        issue_stage(ibuf ^ 1, kk + 32);
        asm volatile("s_wait_asynccnt 0x6" ::: "memory");  // stage kk landed
        __syncthreads();
        compute_stage(ibuf);
        __syncthreads();     // all waves done reading buf before re-fill
    }
    asm volatile("s_wait_asynccnt 0x0" ::: "memory");
    __syncthreads();
    compute_stage(ibuf);     // epilogue stage

    // C/D layout: VGPR r -> row 8*half + r, col l15 (within 16x16 tile)
    const int col = n0 + wn * 16 + l15;
    const float vbi = bi[col], vbf = bf_[col], vbo = bo[col], vbs = bs[col];
    #pragma unroll
    for (int sub = 0; sub < 2; ++sub) {
        const v8f& ci = sub ? ci1 : ci0;
        const v8f& cf = sub ? cf1 : cf0;
        const v8f& co = sub ? co1 : co0;
        const v8f& cs = sub ? cs1 : cs0;
        const int mBase = m0 + wm * 32 + sub * 16 + 8 * half;
        #pragma unroll
        for (int r = 0; r < 8; ++r) {
            const size_t off = (size_t)(mBase + r) * DIM + col;
            float iv = sigm(ci[r] + vbi);
            float fv = sigm(cf[r] + vbf);
            float ov = sigm(co[r] + vbo);
            float sv = tanh_fast(cs[r] + vbs);
            Fg[off] = fv;
            Ag[off] = iv * sv;
            Og[off] = ov;
        }
    }
}

// ---- 4) chunked parallel linear recurrence ---------------------------------
__global__ void scan_part1_kernel(const float* __restrict__ Fg,
                                  const float* __restrict__ Ag,
                                  float* __restrict__ P,
                                  float* __restrict__ Q) {
    const int idx = blockIdx.x * blockDim.x + threadIdx.x; // 0..B*D*CHUNK-1
    const int d = idx & (DIM - 1);
    const int c = (idx >> 10) & (CHUNK - 1);
    const int b = idx >> 15;
    size_t base = (((size_t)b * SEQ) + (size_t)c * CLEN) * DIM + d;
    float p = 1.0f, q = 0.0f;
    for (int tt = 0; tt < CLEN; ++tt) {
        const size_t o = base + (size_t)tt * DIM;
        const float f = Fg[o], a = Ag[o];
        p *= f;
        q = f * q + a;
    }
    P[idx] = p;   // layout (b, c, d), d fastest -> coalesced
    Q[idx] = q;
}

__global__ void scan_part2_kernel(const float* __restrict__ P,
                                  const float* __restrict__ Q,
                                  float* __restrict__ H0) {
    const int idx = blockIdx.x * blockDim.x + threadIdx.x;  // 0..B*D-1
    const int d = idx & (DIM - 1);
    const int b = idx >> 10;
    float h = 0.0f;
    #pragma unroll
    for (int c = 0; c < CHUNK; ++c) {
        const size_t o = (((size_t)b * CHUNK) + c) * DIM + d;
        H0[o] = h;                      // h entering chunk c
        h = P[o] * h + Q[o];
    }
}

__global__ void scan_part3_kernel(const float* __restrict__ Fg,
                                  const float* __restrict__ Ag,
                                  const float* __restrict__ Og,
                                  const float* __restrict__ x,
                                  const float* __restrict__ H0,
                                  float* __restrict__ z) {
    const int idx = blockIdx.x * blockDim.x + threadIdx.x;
    const int d = idx & (DIM - 1);
    const int c = (idx >> 10) & (CHUNK - 1);
    const int b = idx >> 15;
    size_t base = (((size_t)b * SEQ) + (size_t)c * CLEN) * DIM + d;
    float h = H0[idx];
    for (int tt = 0; tt < CLEN; ++tt) {
        const size_t o = base + (size_t)tt * DIM;
        h = Fg[o] * h + Ag[o];
        z[o] = x[o] + Og[o] * h;
    }
}

// ---- 5) LayerNorm over last dim (in-place on d_out) ------------------------
__global__ void __launch_bounds__(256)
layernorm_kernel(float* __restrict__ z,
                 const float* __restrict__ gamma,
                 const float* __restrict__ beta) {
    const size_t base = (size_t)blockIdx.x * DIM;
    const int t = threadIdx.x;
    float v[4];
    float s = 0.0f, q = 0.0f;
    #pragma unroll
    for (int j = 0; j < 4; ++j) {
        v[j] = z[base + t + 256 * j];
        s += v[j];
        q += v[j] * v[j];
    }
    __shared__ float rs[256], rq[256];
    rs[t] = s; rq[t] = q;
    __syncthreads();
    for (int off = 128; off > 0; off >>= 1) {
        if (t < off) { rs[t] += rs[t + off]; rq[t] += rq[t + off]; }
        __syncthreads();
    }
    const float mu  = rs[0] * (1.0f / DIM);
    const float var = rq[0] * (1.0f / DIM) - mu * mu;
    const float inv = rsqrtf(var + 1e-5f);
    #pragma unroll
    for (int j = 0; j < 4; ++j) {
        const int c = t + 256 * j;
        z[base + c] = (v[j] - mu) * inv * gamma[c] + beta[c];
    }
}

// ---------------------------------------------------------------------------
extern "C" void kernel_launch(void* const* d_in, const int* in_sizes, int n_in,
                              void* d_out, int out_size, void* d_ws, size_t ws_size,
                              hipStream_t stream) {
    (void)in_sizes; (void)n_in; (void)out_size; (void)ws_size;

    const float* x     = (const float*)d_in[0];
    const float* Wi    = (const float*)d_in[1];
    const float* bi    = (const float*)d_in[2];
    const float* Wf    = (const float*)d_in[3];
    const float* bf    = (const float*)d_in[4];
    const float* Wo    = (const float*)d_in[5];
    const float* bo    = (const float*)d_in[6];
    const float* Ws    = (const float*)d_in[7];
    const float* bs    = (const float*)d_in[8];
    const float* gamma = (const float*)d_in[9];
    const float* beta  = (const float*)d_in[10];
    float* out = (float*)d_out;

    // Workspace layout (~235 MB):
    //   xb: 32MB | Wt x4: 8MB | Fg/Ag/Og: 192MB | P/Q/H0: 3MB
    char* ws = (char*)d_ws;
    const size_t szXb = (size_t)MROWS * DIM * 2;
    const size_t szWt = (size_t)DIM * DIM * 2;
    const size_t szG  = (size_t)MROWS * DIM * 4;
    const size_t szPQ = (size_t)BATCH * DIM * CHUNK * 4;
    unsigned short* xb  = (unsigned short*)ws;
    unsigned short* Wti = (unsigned short*)(ws + szXb);
    unsigned short* Wtf = (unsigned short*)(ws + szXb + 1 * szWt);
    unsigned short* Wto = (unsigned short*)(ws + szXb + 2 * szWt);
    unsigned short* Wts = (unsigned short*)(ws + szXb + 3 * szWt);
    float* Fg = (float*)(ws + szXb + 4 * szWt);
    float* Ag = (float*)(ws + szXb + 4 * szWt + szG);
    float* Og = (float*)(ws + szXb + 4 * szWt + 2 * szG);
    float* P  = (float*)(ws + szXb + 4 * szWt + 3 * szG);
    float* Q  = (float*)(ws + szXb + 4 * szWt + 3 * szG + szPQ);
    float* H0 = (float*)(ws + szXb + 4 * szWt + 3 * szG + 2 * szPQ);

    // 1) x -> bf16
    conv_x_kernel<<<(MROWS * DIM / 4) / 256, 256, 0, stream>>>(x, xb);

    // 2) W -> W^T bf16 (all four gates)
    {
        dim3 grid(DIM / 32, DIM / 32, 4), block(32, 8);
        transpose_conv_w_kernel<<<grid, block, 0, stream>>>(Wi, Wf, Wo, Ws,
                                                            Wti, Wtf, Wto, Wts);
    }
    // 3) fused 4-gate WMMA GEMM + activations (async-LDS double buffered)
    {
        dim3 grid(DIM / 64, MROWS / 64);
        gates_gemm_kernel<<<grid, 256, 0, stream>>>(xb, Wti, Wtf, Wto, Wts,
                                                    bi, bf, bo, bs, Fg, Ag, Og);
    }
    // 4) chunked parallel scan -> z into d_out
    {
        const int n1 = BATCH * DIM * CHUNK;          // 262144
        scan_part1_kernel<<<n1 / 256, 256, 0, stream>>>(Fg, Ag, P, Q);
        scan_part2_kernel<<<(BATCH * DIM) / 256, 256, 0, stream>>>(P, Q, H0);
        scan_part3_kernel<<<n1 / 256, 256, 0, stream>>>(Fg, Ag, Og, x, H0, out);
    }
    // 5) LayerNorm in-place on d_out
    layernorm_kernel<<<MROWS, 256, 0, stream>>>(out, gamma, beta);
}
